// DecoderAttn_1606317769115
// MI455X (gfx1250) — matrix operand order repaired and checked by reading, not compiled, not run on hardware
//
#include <hip/hip_runtime.h>
#include <hip/hip_bf16.h>
#include <math.h>

// ---------------------------------------------------------------------------
// Problem constants: B=64, T=1024, H=256, L=1
// ---------------------------------------------------------------------------
constexpr int kB = 64;
constexpr int kT = 1024;
constexpr int kH = 256;
constexpr int kM = kT * kB;          // 65536 flattened (t,b) rows

typedef __bf16 bf16;
typedef __attribute__((ext_vector_type(16))) __bf16 v16bf;
typedef __attribute__((ext_vector_type(8)))  __bf16 v8bf;
typedef __attribute__((ext_vector_type(8)))  float  v8f;
typedef __attribute__((ext_vector_type(4)))  unsigned int u32x4;
typedef __attribute__((ext_vector_type(8)))  int i32x8;
typedef __attribute__((ext_vector_type(4)))  int i32x4;

// ---------------------------------------------------------------------------
// Branch-free transcendentals (native v_exp_f32 / v_rcp_f32; the OCML tanhf
// expansion is branchy with exec-mask saves -- bad inside the sequential
// recurrence loop).
// ---------------------------------------------------------------------------
__device__ __forceinline__ float fast_tanh(float x) {
  const float xc = fminf(fmaxf(x, -15.f), 15.f);   // keep exp finite
  const float e  = __expf(2.f * xc);
  return (e - 1.f) * __builtin_amdgcn_rcpf(e + 1.f);
}
__device__ __forceinline__ float fast_sigmoid(float x) {
  return __builtin_amdgcn_rcpf(1.f + __expf(-x));  // saturates cleanly at 0/1
}

// ---------------------------------------------------------------------------
// WMMA helpers (CDNA5 wave32, 16x16x32 bf16 -> f32)
// A layout (16x32): lane m=lane&15, half=lane>>4; K runs [k0+8h..+7] and
//                   [k0+16+8h..+7] (two contiguous b128 loads)
// B layout (32x16): pre-packed so each lane reads 16 contiguous bf16
// C layout (16x16 f32): VGPR r -> M = r + 8*(lane>>4), N = lane&15
// ---------------------------------------------------------------------------
__device__ __forceinline__ v8f wmma_bf16(v16bf a, v16bf b, v8f c) {
  return __builtin_amdgcn_wmma_f32_16x16x32_bf16(
      /*neg_a=*/false, a, /*neg_b=*/false, b,
      /*c_mod=*/(short)0, c, /*reuse_a=*/false, /*reuse_b=*/false);
}

__device__ __forceinline__ v16bf load_a_frag(const bf16* __restrict__ base,
                                             int row0, int rstride, int k0,
                                             int lane) {
  const int half = lane >> 4;
  const int r    = lane & 15;
  const bf16* p  = base + (size_t)(row0 + r) * rstride + k0 + 8 * half;
  v8bf lo = *(const v8bf*)p;
  v8bf hi = *(const v8bf*)(p + 16);
  v16bf a;
#pragma unroll
  for (int i = 0; i < 8; ++i) { a[i] = lo[i]; a[8 + i] = hi[i]; }
  return a;
}

__device__ __forceinline__ v16bf load_b_frag(const bf16* __restrict__ wp,
                                             int nt, int kt, int lane) {
  const bf16* p = wp + ((((nt << 3) + kt) << 5) + lane) * 16;
  v8bf lo = *(const v8bf*)p;
  v8bf hi = *(const v8bf*)(p + 8);
  v16bf b;
#pragma unroll
  for (int i = 0; i < 8; ++i) { b[i] = lo[i]; b[8 + i] = hi[i]; }
  return b;
}

// ---------------------------------------------------------------------------
// TDM staging of a contiguous 32 KB tile into LDS (CDNA5 ISA
// section 8: D# group0 = {count|flags, lds_addr, global_addr lo, global_addr
// hi | type=2}; group1 packs data_size=4B, tensor_dim0=tile_dim0=8192,
// dim1=1, stride0=8192). This toolchain exposes the 6-arg builtin
// (g0, g1, g2, g3, extra group words, cpol); unused groups zero-filled.
// Guarded: falls back to a synchronous vector copy if unavailable.
// ---------------------------------------------------------------------------
#define ATHENA_HAVE_TDM                                            \
  (__has_builtin(__builtin_amdgcn_tensor_load_to_lds) &&           \
   __has_builtin(__builtin_amdgcn_s_wait_tensorcnt))

__device__ __forceinline__ void stage_tile_32k(const bf16* __restrict__ gsrc,
                                               bf16* lds_dst, int tid,
                                               int nthreads, int wave) {
#if ATHENA_HAVE_TDM
  const unsigned long long ga = (unsigned long long)gsrc;
  const unsigned lds = (unsigned)(unsigned long long)(void*)lds_dst;
  u32x4 g0;
  g0[0] = 1u;                                        // count=1, user mode
  g0[1] = lds;                                       // lds_addr (bytes)
  g0[2] = (unsigned)(ga & 0xffffffffu);              // global_addr[31:0]
  g0[3] = (unsigned)((ga >> 32) & 0x1ffffffu) | (2u << 30);  // [56:32]|type=2
  i32x8 g1;
  g1[0] = (int)(2u << 16);          // workgroup_mask=0, data_size=2 (4B)
  g1[1] = (int)(8192u << 16);       // tensor_dim0[15:0] in bits 63:48
  g1[2] = (int)(1u << 16);          // tensor_dim0 hi=0, tensor_dim1 lo16=1
  g1[3] = (int)(8192u << 16);       // tensor_dim1 hi=0, tile_dim0=8192
  g1[4] = 1;                        // tile_dim1=1, tile_dim2=0
  g1[5] = 8192;                     // tensor_dim0_stride lo32
  g1[6] = 0;
  g1[7] = 0;
  i32x4 gz4 = {0, 0, 0, 0};
  i32x8 gz8 = {0, 0, 0, 0, 0, 0, 0, 0};
  if (wave == 0)                    // one DMA per workgroup (EXEC ignored)
    __builtin_amdgcn_tensor_load_to_lds(g0, g1, gz4, gz4, gz8, 0);
  __builtin_amdgcn_s_wait_tensorcnt(0);
  (void)tid; (void)nthreads;
#else
  for (int i = tid; i < 64 * kH / 8; i += nthreads)
    ((v8bf*)lds_dst)[i] = ((const v8bf*)gsrc)[i];
  (void)wave;
#endif
}

// ---------------------------------------------------------------------------
// Weight packer: W[n][h] (f32, row stride `rstride`, col offset `coff`)
//   -> bf16 B-fragment order: dst[((nt*8+kt)*32+lane)*16 + e],
//      lane = 16*((hh>>3)&1) + (n&15), e = (hh&7) + 8*(hh>>4), hh = h&31
// ---------------------------------------------------------------------------
__global__ void pack_weight_kernel(const float* __restrict__ src, int rstride,
                                   int coff, bf16* __restrict__ dst) {
  const int n = blockIdx.x;          // 256 blocks
  const int h = threadIdx.x;         // 256 threads
  const float v  = src[(size_t)n * rstride + coff + h];
  const int nt   = n >> 4, nn = n & 15;
  const int kt   = h >> 5;
  const int hh   = h & 31;
  const int lane = (((hh >> 3) & 1) << 4) + nn;
  const int e    = (hh & 7) + ((hh >> 4) << 3);
  dst[(((size_t)(nt * 8 + kt) * 32 + lane) * 16) + e] = (bf16)v;
}

// inputs (b,t,h) f32  ->  inA[(t*64+b)][h] bf16 (GEMM A rows contiguous)
__global__ void pack_inputs_kernel(const float* __restrict__ inp,
                                   bf16* __restrict__ inA) {
  const int m = blockIdx.x;          // 65536 blocks
  const int h = threadIdx.x;         // 256 threads
  const int b = m & 63, t = m >> 6;
  inA[(size_t)m * kH + h] = (bf16)inp[((size_t)b * kT + t) * kH + h];
}

// hp[b][k] = sum_h hs0[b][h] * W_w[k][h]   (64x256x256 -- tiny, VALU)
__global__ void hp_kernel(const float* __restrict__ hs0,
                          const float* __restrict__ W_w,
                          float* __restrict__ hp) {
  __shared__ float hb[kH];
  const int b = blockIdx.x, k = threadIdx.x;
  hb[k] = hs0[(size_t)b * kH + k];
  __syncthreads();
  const float* wr = W_w + (size_t)k * (2 * kH);   // first half of row k
  float acc = 0.f;
#pragma unroll 8
  for (int h = 0; h < kH; ++h) acc = fmaf(hb[h], wr[h], acc);
  hp[(size_t)b * kH + k] = acc;
}

// ---------------------------------------------------------------------------
// Fused attention: xp = A@W_inp^T ; z = tanh(xp + hp + W_b) (LDS, bf16)
//                  scores = z@V_w^T + V_b   -> scores[b][t][k]
// 4 waves / block, 16 rows / wave, one t per block (m0 multiple of 64)
// ---------------------------------------------------------------------------
__global__ __launch_bounds__(128) void attn_scores_kernel(
    const bf16* __restrict__ inA, const bf16* __restrict__ wB_inp,
    const bf16* __restrict__ wB_vw, const float* __restrict__ hp,
    const float* __restrict__ W_b, const float* __restrict__ V_b,
    float* __restrict__ scores) {
  __shared__ bf16 zbuf[64 * kH];     // 32 KB
  const int lane = threadIdx.x & 31;
  const int wave = threadIdx.x >> 5;
  const int m0   = blockIdx.x * 64 + wave * 16;
  const int t    = m0 >> 6;
  const int half = lane >> 4, nn = lane & 15;

  v16bf afr[8];
#pragma unroll
  for (int kt = 0; kt < 8; ++kt) afr[kt] = load_a_frag(inA, m0, kH, kt * 32, lane);

  // ---- GEMM1 + tanh epilogue into LDS ----
  for (int nt = 0; nt < 16; ++nt) {
    v8f c = {};
#pragma unroll
    for (int kt = 0; kt < 8; ++kt) {
      v16bf bfg = load_b_frag(wB_inp, nt, kt, lane);
      c = wmma_bf16(afr[kt], bfg, c);
    }
    const int n = nt * 16 + nn;
    const float wb = W_b[n];
#pragma unroll
    for (int r = 0; r < 8; ++r) {
      const int row = r + 8 * half;                  // 0..15 within wave tile
      const int b   = (m0 + row) & 63;
      const float z = fast_tanh(c[r] + hp[(size_t)b * kH + n] + wb);
      zbuf[(size_t)(wave * 16 + row) * kH + n] = (bf16)z;
    }
  }
  __syncthreads();

  // ---- GEMM2: scores ----
  v16bf zfr[8];
#pragma unroll
  for (int kt = 0; kt < 8; ++kt)
    zfr[kt] = load_a_frag(zbuf, wave * 16, kH, kt * 32, lane);

  for (int nt = 0; nt < 16; ++nt) {
    v8f c = {};
#pragma unroll
    for (int kt = 0; kt < 8; ++kt) {
      v16bf bfg = load_b_frag(wB_vw, nt, kt, lane);
      c = wmma_bf16(zfr[kt], bfg, c);
    }
    const int n = nt * 16 + nn;
    const float vb = V_b[n];                         // + V_b * L, L==1
#pragma unroll
    for (int r = 0; r < 8; ++r) {
      const int row = r + 8 * half;
      const int b   = (m0 + row) & 63;
      scores[((size_t)b * kT + t) * kH + n] = c[r] + vb;
    }
  }
}

// ---------------------------------------------------------------------------
// Softmax over t per (b,k); ctx = scores*sm -> ctxT[(t*64+b)][k] bf16;
// x_attn[b][t] = sum_k sm  (wave shuffle reduction, no atomics)
// 1 block per b, 256 threads (8 waves)
// ---------------------------------------------------------------------------
__global__ __launch_bounds__(256) void softmax_kernel(
    const float* __restrict__ scores, bf16* __restrict__ ctxT,
    float* __restrict__ x_attn) {
  __shared__ float shM[kH];
  __shared__ float shS[kH];
  const int b = blockIdx.x;
  const int k = threadIdx.x;

  // pass 1: online max/sum along t for this (b,k) column
  {
    const float* col = scores + (size_t)b * kT * kH + k;
    float mx = -3.0e38f, s = 0.f;
    for (int t = 0; t < kT; ++t) {
      const float v  = col[(size_t)t * kH];
      const float nm = fmaxf(mx, v);
      s  = s * __expf(mx - nm) + __expf(v - nm);
      mx = nm;
    }
    shM[k] = mx;
    shS[k] = s;
  }
  __syncthreads();

  // pass 2: wave-per-t; lane owns 8 consecutive k
  const int lane = threadIdx.x & 31;
  const int wv   = threadIdx.x >> 5;
  float m8[8], si8[8];
#pragma unroll
  for (int i = 0; i < 8; ++i) {
    m8[i]  = shM[lane * 8 + i];
    si8[i] = __builtin_amdgcn_rcpf(shS[lane * 8 + i]);
  }
  for (int t = wv; t < kT; t += 8) {
    const float* row = scores + ((size_t)b * kT + t) * kH + lane * 8;
    float xs = 0.f;
    v8bf cxv;
#pragma unroll
    for (int i = 0; i < 8; ++i) {
      const float v  = row[i];
      const float sm = __expf(v - m8[i]) * si8[i];
      xs += sm;
      cxv[i] = (bf16)(v * sm);
    }
    *(v8bf*)(ctxT + ((size_t)(t * kB + b)) * kH + lane * 8) = cxv;
#pragma unroll
    for (int off = 16; off > 0; off >>= 1) xs += __shfl_xor(xs, off, 32);
    if (lane == 0) x_attn[(size_t)b * kT + t] = xs;
  }
}

// ---------------------------------------------------------------------------
// Gate pre-activations: xr/xz/xn[(t*64+b)][k] = ctx@w_i{r,z,n}^T + b_i{r,z,n}
// ctx A-tile (32 KB) staged to LDS by the Tensor Data Mover; A-fragments
// read via ds_load_b128, reused across the 3 weight matrices.
// ---------------------------------------------------------------------------
__global__ __launch_bounds__(128) void gates_kernel(
    const bf16* __restrict__ ctxT, const bf16* __restrict__ wB_ir,
    const bf16* __restrict__ wB_iz, const bf16* __restrict__ wB_in,
    const float* __restrict__ b_ir, const float* __restrict__ b_iz,
    const float* __restrict__ b_in, float* __restrict__ xr,
    float* __restrict__ xz, float* __restrict__ xn) {
  __shared__ bf16 atile[64 * kH];    // 32 KB
  const int lane = threadIdx.x & 31;
  const int wave = threadIdx.x >> 5;
  const int m0   = blockIdx.x * 64 + wave * 16;
  const int half = lane >> 4, nn = lane & 15;

  stage_tile_32k(ctxT + (size_t)blockIdx.x * 64 * kH, atile, threadIdx.x, 128,
                 wave);
  __syncthreads();

  v16bf afr[8];
#pragma unroll
  for (int kt = 0; kt < 8; ++kt)
    afr[kt] = load_a_frag(atile, wave * 16, kH, kt * 32, lane);

  for (int nt = 0; nt < 16; ++nt) {
    v8f cr = {}, cz = {}, cn = {};
#pragma unroll
    for (int kt = 0; kt < 8; ++kt) {
      __builtin_prefetch(wB_ir + ((((nt << 3) + kt + 1) << 5) + lane) * 16, 0, 0);
      v16bf br = load_b_frag(wB_ir, nt, kt, lane);
      cr = wmma_bf16(afr[kt], br, cr);
      v16bf bz = load_b_frag(wB_iz, nt, kt, lane);
      cz = wmma_bf16(afr[kt], bz, cz);
      v16bf bn = load_b_frag(wB_in, nt, kt, lane);
      cn = wmma_bf16(afr[kt], bn, cn);
    }
    const int n = nt * 16 + nn;
    const float vr = b_ir[n], vz = b_iz[n], vn = b_in[n];
#pragma unroll
    for (int r = 0; r < 8; ++r) {
      const size_t off = (size_t)(m0 + r + 8 * half) * kH + n;
      xr[off] = cr[r] + vr;
      xz[off] = cz[r] + vz;
      xn[off] = cn[r] + vn;
    }
  }
}

// ---------------------------------------------------------------------------
// GRU recurrence: persistent kernel, 4 blocks x 16 batch rows, 16 waves
// (one 16-wide N-tile each), h double-buffered in LDS (bf16 for WMMA A,
// f32 for the elementwise update), one barrier per timestep.
// ---------------------------------------------------------------------------
__global__ __launch_bounds__(512) void recurrence_kernel(
    const float* __restrict__ hs0, const bf16* __restrict__ wB_hr,
    const bf16* __restrict__ wB_hz, const bf16* __restrict__ wB_hn,
    const float* __restrict__ b_hr, const float* __restrict__ b_hz,
    const float* __restrict__ b_hn, const float* __restrict__ xr,
    const float* __restrict__ xz, const float* __restrict__ xn,
    float* __restrict__ outputs, float* __restrict__ h_last) {
  __shared__ bf16  hb[2][16 * kH];   // 2 x 8 KB
  __shared__ float hf[2][16 * kH];   // 2 x 16 KB
  const int lane  = threadIdx.x & 31;
  const int wave  = threadIdx.x >> 5;      // 0..15 -> N-tile
  const int bbase = blockIdx.x * 16;       // 4 blocks cover B=64
  const int half  = lane >> 4, nn = lane & 15;
  const int nt    = wave;
  const int n     = nt * 16 + nn;

  // init h0; outputs[:,0,:] = h0
  for (int i = threadIdx.x; i < 16 * kH; i += 512) {
    const int row = i >> 8, k = i & 255;
    const float v = hs0[(size_t)(bbase + row) * kH + k];
    hf[0][i] = v;
    hb[0][i] = (bf16)v;
    outputs[((size_t)(bbase + row) * kT + 0) * kH + k] = v;
  }
  __syncthreads();

  const float vhr = b_hr[n], vhz = b_hz[n], vhn = b_hn[n];

  for (int t = 1; t < kT; ++t) {
    const int cur = (t - 1) & 1, nxt = t & 1;
    v8f cr = {}, cz = {}, cn = {};
#pragma unroll
    for (int kt = 0; kt < 8; ++kt) {
      v16bf a  = load_a_frag(hb[cur], 0, kH, kt * 32, lane);
      v16bf br = load_b_frag(wB_hr, nt, kt, lane);
      cr = wmma_bf16(a, br, cr);
      v16bf bz = load_b_frag(wB_hz, nt, kt, lane);
      cz = wmma_bf16(a, bz, cz);
      v16bf bn = load_b_frag(wB_hn, nt, kt, lane);
      cn = wmma_bf16(a, bn, cn);
    }
#pragma unroll
    for (int r = 0; r < 8; ++r) {
      const int row = r + 8 * half;            // batch row within group
      const int bg  = bbase + row;
      const size_t g = ((size_t)t * kB + bg) * kH + n;
      const float xrv = xr[g], xzv = xz[g], xnv = xn[g];
      const float hprev = hf[cur][row * kH + n];
      const float rg = fast_sigmoid(xrv + cr[r] + vhr);
      const float ng = fast_tanh(xnv + rg * (cn[r] + vhn));
      const float zg = fast_sigmoid(xzv + cz[r] + vhz);
      const float hnew = (1.f - zg) * ng + zg * hprev;
      hf[nxt][row * kH + n] = hnew;
      hb[nxt][row * kH + n] = (bf16)hnew;
      outputs[((size_t)bg * kT + t) * kH + n] = hnew;
    }
    __syncthreads();
  }

  for (int i = threadIdx.x; i < 16 * kH; i += 512) {
    const int row = i >> 8, k = i & 255;
    h_last[(size_t)(bbase + row) * kH + k] = hf[1][i];  // (kT-1)&1 == 1
  }
}

// ---------------------------------------------------------------------------
// Host launcher
// ---------------------------------------------------------------------------
extern "C" void kernel_launch(void* const* d_in, const int* in_sizes, int n_in,
                              void* d_out, int out_size, void* d_ws,
                              size_t ws_size, hipStream_t stream) {
  (void)in_sizes; (void)n_in; (void)out_size; (void)ws_size;

  const float* inputs = (const float*)d_in[0];
  const float* hidden = (const float*)d_in[1];
  const float* W_w    = (const float*)d_in[2];
  const float* W_b    = (const float*)d_in[3];
  const float* V_w    = (const float*)d_in[4];
  const float* V_b    = (const float*)d_in[5];
  const float* w_ir   = (const float*)d_in[6];
  const float* w_iz   = (const float*)d_in[7];
  const float* w_in   = (const float*)d_in[8];
  const float* b_ir   = (const float*)d_in[9];
  const float* b_iz   = (const float*)d_in[10];
  const float* b_in   = (const float*)d_in[11];
  const float* w_hr   = (const float*)d_in[12];
  const float* w_hz   = (const float*)d_in[13];
  const float* w_hn   = (const float*)d_in[14];
  const float* b_hr   = (const float*)d_in[15];
  const float* b_hz   = (const float*)d_in[16];
  const float* b_hn   = (const float*)d_in[17];

  float* out_outputs = (float*)d_out;                       // (64,1024,256)
  float* out_hlast   = out_outputs + (size_t)kB * kT * kH;  // (1,64,256)
  float* out_xattn   = out_hlast + (size_t)kB * kH;         // (64,1024)

  // Workspace layout (overlaid regions; all regions 256B-aligned):
  char* ws = (char*)d_ws;
  const size_t SZ_INA    = (size_t)kM * kH * sizeof(bf16);    // 32 MB (inA, later ctxT)
  const size_t SZ_SCORES = (size_t)kM * kH * sizeof(float);   // 64 MB (scores, later xr)
  bf16*  inA    = (bf16*)ws;                 bf16* ctxT = inA;
  float* scores = (float*)(ws + SZ_INA);     float* xr  = scores;
  float* xz     = (float*)(ws + SZ_INA + 1 * SZ_SCORES);
  float* xn     = (float*)(ws + SZ_INA + 2 * SZ_SCORES);
  float* hp     = (float*)(ws + SZ_INA + 3 * SZ_SCORES);
  bf16*  wpack  = (bf16*)(ws + SZ_INA + 3 * SZ_SCORES + (size_t)kB * kH * sizeof(float));
  const size_t WSZ = (size_t)kH * kH;        // elements per packed matrix
  bf16* wpInp = wpack + 0 * WSZ;
  bf16* wpVw  = wpack + 1 * WSZ;
  bf16* wpIr  = wpack + 2 * WSZ;
  bf16* wpIz  = wpack + 3 * WSZ;
  bf16* wpIn  = wpack + 4 * WSZ;
  bf16* wpHr  = wpack + 5 * WSZ;
  bf16* wpHz  = wpack + 6 * WSZ;
  bf16* wpHn  = wpack + 7 * WSZ;

  // --- stage 0: pack weights + inputs, compute hp ---
  pack_weight_kernel<<<kH, kH, 0, stream>>>(W_w, 2 * kH, kH, wpInp); // W_inp = W_w[:, H:]
  pack_weight_kernel<<<kH, kH, 0, stream>>>(V_w, kH, 0, wpVw);
  pack_weight_kernel<<<kH, kH, 0, stream>>>(w_ir, kH, 0, wpIr);
  pack_weight_kernel<<<kH, kH, 0, stream>>>(w_iz, kH, 0, wpIz);
  pack_weight_kernel<<<kH, kH, 0, stream>>>(w_in, kH, 0, wpIn);
  pack_weight_kernel<<<kH, kH, 0, stream>>>(w_hr, kH, 0, wpHr);
  pack_weight_kernel<<<kH, kH, 0, stream>>>(w_hz, kH, 0, wpHz);
  pack_weight_kernel<<<kH, kH, 0, stream>>>(w_hn, kH, 0, wpHn);
  pack_inputs_kernel<<<kM, kH, 0, stream>>>(inputs, inA);
  hp_kernel<<<kB, kH, 0, stream>>>(hidden, W_w, hp);

  // --- stage 1: fused xp -> tanh(z) -> scores ---
  attn_scores_kernel<<<kM / 64, 128, 0, stream>>>(inA, wpInp, wpVw, hp, W_b,
                                                  V_b, scores);

  // --- stage 2: softmax over T, ctx, x_attn ---
  softmax_kernel<<<kB, kH, 0, stream>>>(scores, ctxT, out_xattn);

  // --- stage 3: gate pre-activations (3 GEMMs, shared A, TDM-staged) ---
  gates_kernel<<<kM / 64, 128, 0, stream>>>(ctxT, wpIr, wpIz, wpIn, b_ir, b_iz,
                                            b_in, xr, xz, xn);

  // --- stage 4: sequential GRU recurrence (persistent, WMMA per step) ---
  recurrence_kernel<<<kB / 16, 512, 0, stream>>>(hidden, wpHr, wpHz, wpHn,
                                                 b_hr, b_hz, b_hn, xr, xz, xn,
                                                 out_outputs, out_hlast);
}
